// TransformerEncoder_28149215658500
// MI455X (gfx1250) — compile-verified, matrix operand order checked by reading, not run on hardware
//
#include <hip/hip_runtime.h>
#include <hip/hip_bf16.h>

typedef __attribute__((ext_vector_type(16))) _Float16 v16h;
typedef __attribute__((ext_vector_type(8)))  _Float16 v8h;
typedef __attribute__((ext_vector_type(8)))  float    v8f;

#define BQ    16
#define NN    3136        // 56*56 tokens per image
#define DD    256
#define HEADS 8
#define HD    32
#define INNER 256
#define WIN2  49
#define NW    64
#define FFD   1024
#define CPBH  512
#define BN    (BQ * NN)   // 50176 rows

// ---------------------------------------------------------------------------
// helpers
// ---------------------------------------------------------------------------
__device__ __forceinline__ v16h ldtile(const _Float16* p) {
  // WMMA A/B operand fragment: elements 0..7 = K base+0..7, 8..15 = K base+16..23
  v8h lo = *(const v8h*)p;
  v8h hi = *(const v8h*)(p + 16);
  return __builtin_shufflevector(lo, hi, 0, 1, 2, 3, 4, 5, 6, 7,
                                 8, 9, 10, 11, 12, 13, 14, 15);
}

__device__ __forceinline__ v8f wmma16(v16h a, v16h b, v8f c) {
  return __builtin_amdgcn_wmma_f32_16x16x32_f16(false, a, false, b,
                                                (short)0, c, false, false);
}

// ---------------------------------------------------------------------------
// weight transpose + fp32 -> fp16:  Wt[n*K + k] = W[k*N + n]
// ---------------------------------------------------------------------------
__global__ __launch_bounds__(256)
void wt_cvt_kernel(const float* __restrict__ W, _Float16* __restrict__ Wt,
                   int K, int N) {
  int t = blockIdx.x * 256 + threadIdx.x;
  if (t < K * N) {
    int k = t / N, n = t % N;
    Wt[(size_t)n * K + k] = (_Float16)W[t];
  }
}

// xm = x + spe ; xh = f16(xm)
__global__ __launch_bounds__(256)
void add_spe_kernel(const float* __restrict__ x, const float* __restrict__ spe,
                    float* __restrict__ xm, _Float16* __restrict__ xh) {
  size_t i = (size_t)blockIdx.x * 256 + threadIdx.x;
  float v = x[i] + spe[i];
  xm[i] = v;
  xh[i] = (_Float16)v;
}

// ---------------------------------------------------------------------------
// continuous position bias MLP: rel(2) -> 512 relu -> heads, per (i,j) pair
// ---------------------------------------------------------------------------
__global__ __launch_bounds__(64)
void cpb_kernel(const float* __restrict__ cw1, const float* __restrict__ cb1,
                const float* __restrict__ cw2, const float* __restrict__ cb2,
                float* __restrict__ cpb) {
  int t = blockIdx.x * 64 + threadIdx.x;
  if (t >= WIN2 * WIN2) return;
  int i = t / WIN2, j = t % WIN2;
  float di = (float)(j / 7 - i / 7);
  float dj = (float)(j % 7 - i % 7);
  float rx = copysignf(log1pf(fabsf(di)), di);
  float ry = copysignf(log1pf(fabsf(dj)), dj);
  float acc[HEADS];
#pragma unroll
  for (int h = 0; h < HEADS; ++h) acc[h] = cb2[h];
  for (int c = 0; c < CPBH; ++c) {
    float hdn = fmaxf(rx * cw1[c] + ry * cw1[CPBH + c] + cb1[c], 0.0f);
#pragma unroll
    for (int h = 0; h < HEADS; ++h) acc[h] += hdn * cw2[c * HEADS + h];
  }
#pragma unroll
  for (int h = 0; h < HEADS; ++h) cpb[h * (WIN2 * WIN2) + t] = acc[h];
}

// ---------------------------------------------------------------------------
// WMMA GEMM with LDS-staged B panel:
//   C[M,N] = A[M,K](f16) @ Wt[N,K](f16)^T   (+bias, +gelu)
// block = 8 waves covering 256 rows x 64 cols; each wave 32 rows.
// Per K-step: shared 64x32 B panel staged to LDS once (4 KB, 1 b128/thread),
// each wave does 8 WMMAs from 2 A fragments x 4 LDS B fragments.
// ---------------------------------------------------------------------------
template <bool GELU, bool OUT16>
__global__ __launch_bounds__(256)
void gemm_wmma_kernel(const _Float16* __restrict__ A,
                      const _Float16* __restrict__ Bt,
                      const float* __restrict__ bias,
                      float* __restrict__ C32, _Float16* __restrict__ C16,
                      int M, int N, int K) {
  const int tid  = threadIdx.x;
  const int lane = tid & 31;
  const int wave = tid >> 5;
  const int ml = lane & 15, hl = lane >> 4;
  const int m0 = blockIdx.y * 256 + wave * 32;
  const int n0 = blockIdx.x * 64;

  __shared__ __align__(16) _Float16 Bs[64 * 32];  // 64 cols x 32 K-halves

  const _Float16* ap0 = A + (size_t)(m0 + ml) * K + hl * 8;
  const _Float16* ap1 = ap0 + (size_t)16 * K;

  // cooperative staging: thread -> (row 0..63, 8-half chunk 0..3)
  const int sr = tid >> 2;
  const int scol = (tid & 3) * 8;
  const _Float16* sp = Bt + (size_t)(n0 + sr) * K + scol;
  _Float16* sdst = Bs + sr * 32 + scol;

  v8h stv = *(const v8h*)sp;        // panel k0 = 0
  *(v8h*)sdst = stv;
  __syncthreads();

  const v8f vzero = {0.f, 0.f, 0.f, 0.f, 0.f, 0.f, 0.f, 0.f};
  v8f acc[2][4];
#pragma unroll
  for (int i = 0; i < 2; ++i)
#pragma unroll
    for (int j = 0; j < 4; ++j) acc[i][j] = vzero;

  for (int k0 = 0; k0 < K; k0 += 32) {
    const bool hasNext = (k0 + 32) < K;
    if (hasNext) {
      stv = *(const v8h*)(sp + k0 + 32);          // next panel -> regs
      __builtin_prefetch(ap0 + k0 + 96, 0, 0);    // global_prefetch_b8
    }
    const v16h a0 = ldtile(ap0 + k0);
    const v16h a1 = ldtile(ap1 + k0);
#pragma unroll
    for (int j = 0; j < 4; ++j) {
      const v16h b = ldtile(Bs + (j * 16 + ml) * 32 + hl * 8);
      acc[0][j] = wmma16(a0, b, acc[0][j]);
      acc[1][j] = wmma16(a1, b, acc[1][j]);
    }
    if (hasNext) {
      __syncthreads();            // all waves done reading current panel
      *(v8h*)sdst = stv;          // overwrite with next panel
      __syncthreads();
    }
  }

#pragma unroll
  for (int i = 0; i < 2; ++i) {
#pragma unroll
    for (int j = 0; j < 4; ++j) {
      const int n = n0 + j * 16 + ml;
      const float bs = bias ? bias[n] : 0.0f;
#pragma unroll
      for (int r = 0; r < 8; ++r) {
        const int m = m0 + i * 16 + r + 8 * hl;
        float val = acc[i][j][r] + bs;
        if (GELU) val = 0.5f * val * (1.0f + erff(val * 0.70710678118f));
        if (OUT16) C16[(size_t)m * N + n] = (_Float16)val;
        else       C32[(size_t)m * N + n] = val;
      }
    }
  }
}

// ---------------------------------------------------------------------------
// windowed cosine attention: one wave32 workgroup per (batch, window, head)
// 49 tokens padded to 64 so WMMA runs with full EXEC
// ---------------------------------------------------------------------------
__global__ __launch_bounds__(32)
void attn_kernel(const _Float16* __restrict__ qkv, const float* __restrict__ tau,
                 const float* __restrict__ cpb, _Float16* __restrict__ attnh,
                 int shifted) {
  const int lane = threadIdx.x;
  const int ml = lane & 15, hl = lane >> 4;
  const int bid = blockIdx.x;
  const int h  = bid & 7;
  const int w  = (bid >> 3) & 63;
  const int b  = bid >> 9;
  const int wy = w >> 3, wx = w & 7;

  __shared__ __align__(16) _Float16 qn[64 * 32];  // l2-normed q, zero padded
  __shared__ __align__(16) _Float16 kn[64 * 32];  // l2-normed k
  __shared__ __align__(16) _Float16 vt[32 * 64];  // v transposed [dim][token]
  __shared__ __align__(16) float    sc[64 * 64];  // scores
  __shared__ __align__(16) _Float16 at[64 * 64];  // softmaxed attn, f16

  for (int i = lane; i < 64 * 32; i += 32) {
    qn[i] = (_Float16)0.0f; kn[i] = (_Float16)0.0f; vt[i] = (_Float16)0.0f;
  }
  for (int i = lane; i < 64 * 64; i += 32) at[i] = (_Float16)0.0f;
  __syncthreads();

  // gather window tokens (roll folded into index math) + l2 normalize q,k
  for (int t = 0; t < WIN2; ++t) {
    int y = wy * 7 + t / 7;
    int x = wx * 7 + t % 7;
    if (shifted) { y = (y + 3) % 56; x = (x + 3) % 56; }
    const size_t row = ((size_t)b * NN + y * 56 + x) * 768;
    const float qf = (float)qkv[row +       h * 32 + lane];
    const float kf = (float)qkv[row + 256 + h * 32 + lane];
    const float vf = (float)qkv[row + 512 + h * 32 + lane];
    float qs = qf * qf, ks = kf * kf;
#pragma unroll
    for (int o = 16; o > 0; o >>= 1) {
      qs += __shfl_xor(qs, o, 32);
      ks += __shfl_xor(ks, o, 32);
    }
    const float qi = 1.0f / fmaxf(sqrtf(qs), 1e-12f);
    const float ki = 1.0f / fmaxf(sqrtf(ks), 1e-12f);
    qn[t * 32 + lane] = (_Float16)(qf * qi);
    kn[t * 32 + lane] = (_Float16)(kf * ki);
    vt[lane * 64 + t] = (_Float16)vf;
  }
  __syncthreads();

  // S = qn @ kn^T : 4x4 tiles of 16x16, single K=32 WMMA each
#pragma unroll
  for (int mt = 0; mt < 4; ++mt) {
    const v16h aq = ldtile(qn + (mt * 16 + ml) * 32 + hl * 8);
#pragma unroll
    for (int nt = 0; nt < 4; ++nt) {
      const v16h bk = ldtile(kn + (nt * 16 + ml) * 32 + hl * 8);
      v8f c = {};
      c = wmma16(aq, bk, c);
#pragma unroll
      for (int r = 0; r < 8; ++r)
        sc[(mt * 16 + r + 8 * hl) * 64 + nt * 16 + ml] = c[r];
    }
  }
  __syncthreads();

  // softmax with tau scaling, CPB bias, shift masks
  const float scale = 1.0f / fmaxf(tau[h], 0.01f);
  const bool mUL = shifted && (wy == 7);
  const bool mLR = shifted && (wx == 7);
  const float* cpbh = cpb + h * (WIN2 * WIN2);
  for (int r = lane; r < WIN2; r += 32) {
    float mx = -3.0e38f;
    for (int j = 0; j < WIN2; ++j) {
      float v = sc[r * 64 + j] * scale + cpbh[r * WIN2 + j];
      if (mUL && ((r >= 28) != (j >= 28)))                 v = -1.0e30f;
      if (mLR && (((r % 7) >= 4) != ((j % 7) >= 4)))       v = -1.0e30f;
      sc[r * 64 + j] = v;
      mx = fmaxf(mx, v);
    }
    float sum = 0.0f;
    for (int j = 0; j < WIN2; ++j) sum += __expf(sc[r * 64 + j] - mx);
    const float inv = 1.0f / sum;
    for (int j = 0; j < WIN2; ++j)
      at[r * 64 + j] = (_Float16)(__expf(sc[r * 64 + j] - mx) * inv);
  }
  __syncthreads();

  // O = attn @ v : 4x2 tiles, K = 64 via two WMMAs
#pragma unroll
  for (int mt = 0; mt < 4; ++mt) {
#pragma unroll
    for (int nt = 0; nt < 2; ++nt) {
      v8f c = {};
#pragma unroll
      for (int kt = 0; kt < 2; ++kt) {
        const v16h aa = ldtile(at + (mt * 16 + ml) * 64 + kt * 32 + hl * 8);
        const v16h bv = ldtile(vt + (nt * 16 + ml) * 64 + kt * 32 + hl * 8);
        c = wmma16(aa, bv, c);
      }
#pragma unroll
      for (int r = 0; r < 8; ++r) {
        const int tk = mt * 16 + r + 8 * hl;
        if (tk < WIN2) {
          int y = wy * 7 + tk / 7;
          int x = wx * 7 + tk % 7;
          if (shifted) { y = (y + 3) % 56; x = (x + 3) % 56; }  // roll back
          attnh[((size_t)b * NN + y * 56 + x) * INNER + h * 32 + nt * 16 + ml] =
              (_Float16)c[r];
        }
      }
    }
  }
}

// ---------------------------------------------------------------------------
// LayerNorm(a)*g+b + xm -> xm (and f16 mirror)  /  final LayerNorm -> out
// ---------------------------------------------------------------------------
__global__ __launch_bounds__(256)
void ln_res_kernel(const float* __restrict__ a, const float* __restrict__ g,
                   const float* __restrict__ bvec, float* __restrict__ xm,
                   _Float16* __restrict__ xh) {
  const int t = blockIdx.x, c = threadIdx.x;
  const size_t idx = (size_t)t * DD + c;
  const float v = a[idx];
  __shared__ float red[256];
  red[c] = v; __syncthreads();
  for (int s = 128; s > 0; s >>= 1) { if (c < s) red[c] += red[c + s]; __syncthreads(); }
  const float mean = red[0] * (1.0f / DD);
  __syncthreads();
  const float d = v - mean;
  red[c] = d * d; __syncthreads();
  for (int s = 128; s > 0; s >>= 1) { if (c < s) red[c] += red[c + s]; __syncthreads(); }
  const float var = red[0] * (1.0f / DD);
  const float y = d * rsqrtf(var + 1e-5f) * g[c] + bvec[c] + xm[idx];
  xm[idx] = y;
  xh[idx] = (_Float16)y;
}

__global__ __launch_bounds__(256)
void final_ln_kernel(const float* __restrict__ xm, const float* __restrict__ g,
                     const float* __restrict__ bvec, float* __restrict__ out) {
  const int t = blockIdx.x, c = threadIdx.x;
  const size_t idx = (size_t)t * DD + c;
  const float v = xm[idx];
  __shared__ float red[256];
  red[c] = v; __syncthreads();
  for (int s = 128; s > 0; s >>= 1) { if (c < s) red[c] += red[c + s]; __syncthreads(); }
  const float mean = red[0] * (1.0f / DD);
  __syncthreads();
  const float d = v - mean;
  red[c] = d * d; __syncthreads();
  for (int s = 128; s > 0; s >>= 1) { if (c < s) red[c] += red[c + s]; __syncthreads(); }
  const float var = red[0] * (1.0f / DD);
  out[idx] = d * rsqrtf(var + 1e-5f) * g[c] + bvec[c];
}

// ---------------------------------------------------------------------------
// host-side orchestration (graph-capture safe: launches only)
// ---------------------------------------------------------------------------
extern "C" void kernel_launch(void* const* d_in, const int* in_sizes, int n_in,
                              void* d_out, int out_size, void* d_ws, size_t ws_size,
                              hipStream_t stream) {
  (void)in_sizes; (void)n_in; (void)out_size; (void)ws_size;
  const float* x    = (const float*)d_in[0];
  const float* spe  = (const float*)d_in[1];
  const float* qkvw = (const float*)d_in[2];
  const float* tau  = (const float*)d_in[3];
  const float* cw1  = (const float*)d_in[4];
  const float* cb1  = (const float*)d_in[5];
  const float* cw2  = (const float*)d_in[6];
  const float* cb2  = (const float*)d_in[7];
  const float* ow   = (const float*)d_in[8];
  const float* ob   = (const float*)d_in[9];
  const float* ln1g = (const float*)d_in[10];
  const float* ln1b = (const float*)d_in[11];
  const float* fw1  = (const float*)d_in[12];
  const float* fb1  = (const float*)d_in[13];
  const float* fw2  = (const float*)d_in[14];
  const float* fb2  = (const float*)d_in[15];
  const float* ln2g = (const float*)d_in[16];
  const float* ln2b = (const float*)d_in[17];
  const float* ng   = (const float*)d_in[18];
  const float* nb   = (const float*)d_in[19];
  float* out = (float*)d_out;

  char* p = (char*)d_ws;
  auto take = [&](size_t bytes) {
    char* r = p; p += (bytes + 255) & ~(size_t)255; return r;
  };
  float*    xm     = (float*)   take((size_t)BN * DD * 4);
  _Float16* xh     = (_Float16*)take((size_t)BN * DD * 2);
  _Float16* qkvb   = (_Float16*)take((size_t)BN * 768 * 2);
  _Float16* attnh  = (_Float16*)take((size_t)BN * INNER * 2);
  float*    pout   = (float*)   take((size_t)BN * DD * 4);
  _Float16* hidden = (_Float16*)take((size_t)BN * FFD * 2);
  float*    cpbb   = (float*)   take((size_t)HEADS * WIN2 * WIN2 * 4);
  _Float16* qkvwt  = (_Float16*)take((size_t)2 * 768 * DD * 2);
  _Float16* owt    = (_Float16*)take((size_t)2 * DD * INNER * 2);
  _Float16* fw1t   = (_Float16*)take((size_t)2 * FFD * DD * 2);
  _Float16* fw2t   = (_Float16*)take((size_t)2 * DD * FFD * 2);

  auto cvt = [&](const float* W, _Float16* Wt, int K, int N) {
    int n = K * N;
    wt_cvt_kernel<<<(n + 255) / 256, 256, 0, stream>>>(W, Wt, K, N);
  };
  for (int l = 0; l < 2; ++l) {
    cvt(qkvw + (size_t)l * DD * 768,   qkvwt + (size_t)l * 768 * DD, DD, 768);
    cvt(ow   + (size_t)l * INNER * DD, owt   + (size_t)l * DD * INNER, INNER, DD);
    cvt(fw1  + (size_t)l * DD * FFD,   fw1t  + (size_t)l * FFD * DD, DD, FFD);
    cvt(fw2  + (size_t)l * FFD * DD,   fw2t  + (size_t)l * DD * FFD, FFD, DD);
  }
  add_spe_kernel<<<BN, 256, 0, stream>>>(x, spe, xm, xh);

  for (int l = 0; l < 2; ++l) {
    cpb_kernel<<<(WIN2 * WIN2 + 63) / 64, 64, 0, stream>>>(
        cw1 + l * 2 * CPBH, cb1 + l * CPBH, cw2 + l * CPBH * HEADS,
        cb2 + l * HEADS, cpbb);
    // QKV projection (no bias) -> f16
    gemm_wmma_kernel<false, true><<<dim3(768 / 64, BN / 256), 256, 0, stream>>>(
        xh, qkvwt + (size_t)l * 768 * DD, nullptr, nullptr, qkvb, BN, 768, DD);
    // windowed attention
    attn_kernel<<<BQ * NW * HEADS, 32, 0, stream>>>(
        qkvb, tau + l * HEADS, cpbb, attnh, l & 1);
    // output projection + bias -> f32
    gemm_wmma_kernel<false, false><<<dim3(DD / 64, BN / 256), 256, 0, stream>>>(
        attnh, owt + (size_t)l * DD * INNER, ob + l * DD, pout, nullptr,
        BN, DD, INNER);
    ln_res_kernel<<<BN, 256, 0, stream>>>(pout, ln1g + l * DD, ln1b + l * DD, xm, xh);
    // FF1 + bias + gelu -> f16
    gemm_wmma_kernel<true, true><<<dim3(FFD / 64, BN / 256), 256, 0, stream>>>(
        xh, fw1t + (size_t)l * FFD * DD, fb1 + l * FFD, nullptr, hidden,
        BN, FFD, DD);
    // FF2 + bias -> f32
    gemm_wmma_kernel<false, false><<<dim3(DD / 64, BN / 256), 256, 0, stream>>>(
        hidden, fw2t + (size_t)l * DD * FFD, fb2 + l * DD, pout, nullptr,
        BN, DD, FFD);
    ln_res_kernel<<<BN, 256, 0, stream>>>(pout, ln2g + l * DD, ln2b + l * DD, xm, xh);
  }
  final_ln_kernel<<<BN, 256, 0, stream>>>(xm, ng, nb, out);
}
